// HyenaOperator_29609504539253
// MI455X (gfx1250) — matrix operand order repaired and checked by reading, not compiled, not run on hardware
//
#include <hip/hip_runtime.h>
#include <cstdint>
#include <cstddef>

// ---------------- CDNA5 types ----------------
typedef __bf16 bf16_t;
typedef __attribute__((ext_vector_type(16))) __bf16 v16bf;
typedef __attribute__((ext_vector_type(8)))  float  v8f;
typedef __attribute__((ext_vector_type(4)))  unsigned int v4u;
typedef __attribute__((ext_vector_type(8)))  int v8i;
typedef __attribute__((ext_vector_type(4)))  int v4i;

union Frag16 { v16bf v; uint4 q[2]; };   // 16 bf16 = 32B = two 16B chunks

// problem sizes (fixed by reference)
constexpr int Dm = 1024;     // embedding dim
constexpr int Em = 3072;     // expanded dim
constexpr int Lm = 4096;     // sequence length
constexpr int Bm = 2;        // batch
constexpr int Mrows = Bm * Lm;   // 8192 GEMM rows
constexpr int Crows = Bm * Dm;   // 2048 FIR channels

// GEMM tiling
#define BM 128
#define BN 128
#define BK 32
#define LDK (BK + 8)   // +8 bf16 pad -> 80B row stride; TDM pad reproduces this

__device__ __forceinline__ float silu_f(float x) { return x / (1.0f + __expf(-x)); }

// ============================================================
// Tensor Data Mover: 2D bf16 tile (tile_d0 x tile_d1) from global to LDS,
// with LDS padding of 4 dwords after every 16 dwords (64B row -> 80B stride).
// D# packing per CDNA5 ISA 8.3-8.5; clang-23 6-arg builtin
// (g0 v4u, g1 v8i, g2 v4i, g3 v4i, g4 v8i, cpol i32).
// ============================================================
__device__ __forceinline__ void tdm_load_2d(uint32_t lds_off, const bf16_t* gp,
                                            uint32_t tensor_d0, uint32_t tensor_d1,
                                            uint32_t tile_d0, uint32_t tile_d1,
                                            uint64_t stride0_elems) {
  uint64_t ga = (uint64_t)(uintptr_t)gp;
  v4u g0;
  g0[0] = 1u;                                            // count=1, user D#
  g0[1] = lds_off;                                       // lds_addr (bytes)
  g0[2] = (uint32_t)ga;                                  // global_addr[31:0]
  g0[3] = (uint32_t)((ga >> 32) & 0x01FFFFFFu) | (2u << 30); // addr[56:32] | type=2
  v8i g1;
  g1[0] = (int)((1u << 16)      // data_size = 2 bytes (bf16)
              | (1u << 20)      // pad_enable
              | (3u << 22)      // pad_interval: every 16 dwords (64B)
              | (3u << 25));    // pad_amount: 4 dwords (16B)
  g1[1] = (int)((tensor_d0 & 0xFFFFu) << 16);                       // dim0[15:0]
  g1[2] = (int)(((tensor_d0 >> 16) & 0xFFFFu) | ((tensor_d1 & 0xFFFFu) << 16));
  g1[3] = (int)(((tensor_d1 >> 16) & 0xFFFFu) | ((tile_d0 & 0xFFFFu) << 16));
  g1[4] = (int)(tile_d1 & 0xFFFFu);                                 // tile_dim1, tile_dim2=0
  g1[5] = (int)(uint32_t)(stride0_elems & 0xFFFFFFFFu);             // dim0_stride lo
  g1[6] = (int)(uint32_t)((stride0_elems >> 32) & 0xFFFFu);         // dim0_stride hi
  g1[7] = 0;
  v4i g2 = {0, 0, 0, 0};
  v4i g3 = {0, 0, 0, 0};
  v8i g4 = {0, 0, 0, 0, 0, 0, 0, 0};
  __builtin_amdgcn_tensor_load_to_lds(g0, g1, g2, g3, g4, 0);
}

// ============================================================
// GEMM: C[M,N] = A[M,K](bf16) @ Bt[N,K]^T(bf16) + bias[N]
// TDM double-buffered LDS staging + v_wmma_f32_16x16x32_bf16.
// 256 threads = 8 waves; wave grid 4(M) x 2(N); each wave 32x64 (2x4 tiles).
// ============================================================
template <typename OutT>
__global__ __launch_bounds__(256)
void gemm_bf16_tdm(const bf16_t* __restrict__ A, const bf16_t* __restrict__ Bt,
                   const float* __restrict__ bias, OutT* __restrict__ C,
                   int M, int N, int K) {
  __shared__ __align__(16) bf16_t As[2][BM * LDK];
  __shared__ __align__(16) bf16_t Bs[2][BN * LDK];

  const int tid  = threadIdx.x;
  const int lane = tid & 31;
  const int wave = tid >> 5;
  const int wm   = wave & 3;        // rows wm*32
  const int wn   = wave >> 2;       // cols wn*64
  const int m0   = blockIdx.x * BM;
  const int n0   = blockIdx.y * BN;
  const int nrow = lane & 15;
  const int half = lane >> 4;

  const v8f zero8 = {0.f,0.f,0.f,0.f,0.f,0.f,0.f,0.f};
  v8f acc[2][4];
  for (int mt = 0; mt < 2; ++mt)
    for (int nt = 0; nt < 4; ++nt) acc[mt][nt] = zero8;

  const int niter = K / BK;
  if (wave == 0) {   // DMA prologue: k-step 0 into buffer 0
    tdm_load_2d((uint32_t)(uintptr_t)&As[0][0], A + (size_t)m0 * K,
                (uint32_t)K, (uint32_t)M, BK, BM, (uint64_t)K);
    tdm_load_2d((uint32_t)(uintptr_t)&Bs[0][0], Bt + (size_t)n0 * K,
                (uint32_t)K, (uint32_t)N, BK, BN, (uint64_t)K);
  }

  for (int it = 0; it < niter; ++it) {
    if (wave == 0) __builtin_amdgcn_s_wait_tensorcnt(0);  // buf[it&1] ready
    __syncthreads();
    if (wave == 0 && it + 1 < niter) {                    // prefetch next k-step
      const int k1 = (it + 1) * BK;
      const int nb = (it + 1) & 1;
      tdm_load_2d((uint32_t)(uintptr_t)&As[nb][0], A + (size_t)m0 * K + k1,
                  (uint32_t)K, (uint32_t)M, BK, BM, (uint64_t)K);
      tdm_load_2d((uint32_t)(uintptr_t)&Bs[nb][0], Bt + (size_t)n0 * K + k1,
                  (uint32_t)K, (uint32_t)N, BK, BN, (uint64_t)K);
    }
    const bf16_t* as = &As[it & 1][0];
    const bf16_t* bs = &Bs[it & 1][0];

    Frag16 a[2], b[4];
    #pragma unroll
    for (int mt = 0; mt < 2; ++mt) {
      // A 16x32: lane row = lane%16; K chunks {h*8..+7} and {16+h*8..+7}
      const bf16_t* p = &as[(wm * 32 + mt * 16 + nrow) * LDK];
      a[mt].q[0] = *(const uint4*)(p + half * 8);
      a[mt].q[1] = *(const uint4*)(p + 16 + half * 8);
    }
    #pragma unroll
    for (int nt = 0; nt < 4; ++nt) {
      // B 32x16: lane col = lane%16; K chunks {h*8..+7} and {16+h*8..+7}
      const bf16_t* p = &bs[(wn * 64 + nt * 16 + nrow) * LDK + half * 8];
      b[nt].q[0] = *(const uint4*)(p);
      b[nt].q[1] = *(const uint4*)(p + 16);
    }
    #pragma unroll
    for (int mt = 0; mt < 2; ++mt)
      #pragma unroll
      for (int nt = 0; nt < 4; ++nt)
        acc[mt][nt] = __builtin_amdgcn_wmma_f32_16x16x32_bf16(
            false, a[mt].v, false, b[nt].v, (short)0, acc[mt][nt], false, false);
  }

  __syncthreads();
  // epilogue: C layout lane->col=lane%16, vgpr r -> row r+8*half
  #pragma unroll
  for (int mt = 0; mt < 2; ++mt)
    #pragma unroll
    for (int nt = 0; nt < 4; ++nt) {
      int col = n0 + wn * 64 + nt * 16 + nrow;
      float bv = bias[col];
      #pragma unroll
      for (int r = 0; r < 8; ++r) {
        int row = m0 + wm * 32 + mt * 16 + r + 8 * half;
        C[(size_t)row * N + col] = (OutT)(acc[mt][nt][r] + bv);
      }
    }
}

// ============================================================
// f32 -> bf16 bulk pack (row-major copy), 4 elems/thread
// ============================================================
__global__ __launch_bounds__(256)
void pack_bf16(const float* __restrict__ s, bf16_t* __restrict__ d) {
  size_t i = ((size_t)blockIdx.x * 256 + threadIdx.x) * 4;
  float4 f = *(const float4*)(s + i);
  union { bf16_t h[4]; uint2 u; } p;
  p.h[0] = (bf16_t)f.x; p.h[1] = (bf16_t)f.y;
  p.h[2] = (bf16_t)f.z; p.h[3] = (bf16_t)f.w;
  *(uint2*)(d + i) = p.u;
}

// ============================================================
// f32 [K,N] -> bf16 [N,K] transpose pack (LDS 32x32 tile)
// ============================================================
__global__ __launch_bounds__(256)
void pack_transpose_bf16(const float* __restrict__ s, bf16_t* __restrict__ d,
                         int K, int N) {
  __shared__ float t[32][33];
  const int n0 = blockIdx.x * 32, k0 = blockIdx.y * 32;
  const int tn = threadIdx.x & 31, tr = threadIdx.x >> 5;
  for (int kk = tr; kk < 32; kk += 8)
    t[kk][tn] = s[(size_t)(k0 + kk) * N + n0 + tn];
  __syncthreads();
  for (int nn = tr; nn < 32; nn += 8)
    d[(size_t)(n0 + nn) * K + k0 + tn] = (bf16_t)t[tn][nn];
}

// ============================================================
// Depthwise causal conv (taps l-2,l-1,l) + gate: v1 = s2*s0 (bf16 [C,L]),
// keep s0 (f32 [C,L]).  streams[1] is dead in the reference.
// ============================================================
__global__ __launch_bounds__(256)
void conv_gate(const bf16_t* __restrict__ U, const float* __restrict__ wconv,
               const float* __restrict__ bconv,
               bf16_t* __restrict__ v1, float* __restrict__ s0buf) {
  const int bl = blockIdx.x;          // b*L + l
  const int b  = bl >> 12;            // L = 4096
  const int l  = bl & (Lm - 1);
  #pragma unroll
  for (int k = 0; k < 4; ++k) {
    int d = threadIdx.x + k * 256;
    float s[2];
    int es[2] = { d, 2 * Dm + d };    // stream 0 and stream 2 channels
    #pragma unroll
    for (int t = 0; t < 2; ++t) {
      int e = es[t];
      const float* w = wconv + e * 3;
      float acc = bconv[e] + w[2] * (float)U[((size_t)(b * Lm + l)) * Em + e];
      if (l >= 1) acc += w[1] * (float)U[((size_t)(b * Lm + l - 1)) * Em + e];
      if (l >= 2) acc += w[0] * (float)U[((size_t)(b * Lm + l - 2)) * Em + e];
      s[t] = acc;
    }
    size_t ci = (size_t)(b * Dm + d) * Lm + l;
    v1[ci]    = (bf16_t)(s[1] * s[0]);   // v = streams[2] * streams[0]
    s0buf[ci] = s[0];
  }
}

// ============================================================
// Implicit filter, row 0 only (t=0 -> pos@w1 == 0, exp-decay == 1):
// h = silu(silu(b1) @ w2 + b2) @ w3
// ============================================================
__global__ __launch_bounds__(1024)
void filter_row0(const float* __restrict__ b1, const float* __restrict__ w2,
                 const float* __restrict__ b2, const float* __restrict__ w3,
                 float* __restrict__ h) {
  __shared__ float a1[64], a2[64];
  int t = threadIdx.x;
  if (t < 64) a1[t] = silu_f(b1[t]);
  __syncthreads();
  if (t < 64) {
    float s = b2[t];
    for (int g = 0; g < 64; ++g) s += a1[g] * w2[g * 64 + t];
    a2[t] = silu_f(s);
  }
  __syncthreads();
  float s = 0.f;
  for (int f = 0; f < 64; ++f) s += a2[f] * w3[f * Dm + t];
  h[t] = s;
}

// ============================================================
// Toeplitz B-tiles of h: bt[dix][i][jj] = h[dix*32 + i - jj] (0 if OOB)
// ============================================================
__global__ __launch_bounds__(512)
void build_bt(const float* __restrict__ h, bf16_t* __restrict__ bt) {
  int dix = blockIdx.x;
  int t   = threadIdx.x;
  int i   = t >> 5;
  int jj  = t & 31;
  int m   = dix * 32 + i - jj;
  float v = (m >= 0 && m < Dm) ? h[m] : 0.f;
  bt[dix * 512 + i * 32 + jj] = (bf16_t)v;
}

// ============================================================
// Causal FIR (len-1024, shared filter) as Toeplitz WMMA + bias + s0 gate.
// One wave per 16x16 (channel x l) tile; 33 WMMA accumulations.
// Output z stored [B*L, D] bf16 -> A matrix of GEMM2.
// ============================================================
__global__ __launch_bounds__(256)
void fir_gate(const bf16_t* __restrict__ v1, const float* __restrict__ s0,
              const bf16_t* __restrict__ bt, const float* __restrict__ fb,
              bf16_t* __restrict__ z) {
  const int bx   = blockIdx.x;
  const int cb   = bx >> 5;                 // channel block 0..127
  const int lg   = bx & 31;                 // l-group
  const int wave = threadIdx.x >> 5;
  const int lane = threadIdx.x & 31;
  const int l0   = (lg * 8 + wave) * 16;
  const int nrow = lane & 15;
  const int half = lane >> 4;

  const bf16_t* arow = v1 + (size_t)(cb * 16 + nrow) * Lm;
  const v8f zero8 = {0.f,0.f,0.f,0.f,0.f,0.f,0.f,0.f};
  uint4 z4; z4.x = z4.y = z4.z = z4.w = 0u;
  v8f acc = zero8;

  for (int dix = 0; dix <= 32; ++dix) {
    int j0 = l0 - dix * 32;
    int g0 = j0 + half * 8;                 // A chunk bases (8 bf16 each)
    int g1 = j0 + 16 + half * 8;
    Frag16 a, b;
    a.q[0] = (g0 >= 0) ? *(const uint4*)(arow + g0) : z4;   // zero-pad left edge
    a.q[1] = (g1 >= 0) ? *(const uint4*)(arow + g1) : z4;
    const bf16_t* bp = bt + dix * 512 + nrow * 32 + half * 8;
    b.q[0] = *(const uint4*)(bp);
    b.q[1] = *(const uint4*)(bp + 16);
    acc = __builtin_amdgcn_wmma_f32_16x16x32_bf16(
        false, a.v, false, b.v, (short)0, acc, false, false);
  }

  const int l = l0 + nrow;                  // C col = lane%16
  #pragma unroll
  for (int r = 0; r < 8; ++r) {
    int cc = cb * 16 + r + 8 * half;        // C row = r + 8*half
    int d  = cc & (Dm - 1);
    int b_ = cc >> 10;
    size_t ci = (size_t)cc * Lm + l;
    float v  = (float)v1[ci];
    float y  = acc[r] + v * fb[d];          // fftconv adds v * filt_bias
    z[((size_t)(b_ * Lm + l)) * Dm + d] = (bf16_t)(y * s0[ci]);  // final gate
  }
}

// ============================================================
extern "C" void kernel_launch(void* const* d_in, const int* in_sizes, int n_in,
                              void* d_out, int out_size, void* d_ws, size_t ws_size,
                              hipStream_t stream) {
  (void)in_sizes; (void)n_in; (void)out_size; (void)ws_size;
  const float* x      = (const float*)d_in[0];
  const float* w_in   = (const float*)d_in[1];
  const float* b_in   = (const float*)d_in[2];
  const float* w_conv = (const float*)d_in[3];
  const float* b_conv = (const float*)d_in[4];
  // d_in[5] = w1: multiplied by pos row0 == [0,0,0] -> unused
  const float* b1     = (const float*)d_in[6];
  const float* w2     = (const float*)d_in[7];
  const float* b2     = (const float*)d_in[8];
  const float* w3     = (const float*)d_in[9];
  const float* fb     = (const float*)d_in[10];
  const float* w_out  = (const float*)d_in[11];
  const float* b_out  = (const float*)d_in[12];
  float* out = (float*)d_out;

  // workspace carve-out (256B aligned regions)
  char* ws = (char*)d_ws;
  size_t off = 0;
  auto carve = [&](size_t bytes) -> void* {
    void* p = ws + off;
    off = (off + bytes + 255) & ~(size_t)255;
    return p;
  };
  bf16_t* xb  = (bf16_t*)carve((size_t)Mrows * Dm * 2);        // A of GEMM1
  bf16_t* wit = (bf16_t*)carve((size_t)Em * Dm * 2);           // w_in^T  [E,D]
  bf16_t* wot = (bf16_t*)carve((size_t)Dm * Dm * 2);           // w_out^T [D,D]
  bf16_t* U   = (bf16_t*)carve((size_t)Mrows * Em * 2);        // 50 MB
  bf16_t* v1  = (bf16_t*)carve((size_t)Crows * Lm * 2 + 256);  // +pad for edge reads
  float*  s0b = (float*) carve((size_t)Crows * Lm * sizeof(float));
  bf16_t* zb  = (bf16_t*)carve((size_t)Mrows * Dm * 2);        // A of GEMM2
  float*  h   = (float*) carve((size_t)Dm * sizeof(float));
  bf16_t* bt  = (bf16_t*)carve((size_t)33 * 512 * 2);

  // 0) one-shot bf16 packs (x, w_in^T, w_out^T)
  pack_bf16<<<(Mrows * Dm) / 1024, 256, 0, stream>>>(x, xb);
  dim3 gt1(Em / 32, Dm / 32);
  pack_transpose_bf16<<<gt1, 256, 0, stream>>>(w_in, wit, Dm, Em);
  dim3 gt2(Dm / 32, Dm / 32);
  pack_transpose_bf16<<<gt2, 256, 0, stream>>>(w_out, wot, Dm, Dm);

  // 1) u = x @ w_in + b_in   (M=8192, N=3072, K=1024) -> bf16 U
  dim3 g1(Mrows / BM, Em / BN);
  gemm_bf16_tdm<bf16_t><<<g1, 256, 0, stream>>>(xb, wit, b_in, U, Mrows, Em, Dm);

  // 2) depthwise causal conv + gate -> v1 (bf16 [C,L]), s0 (f32 [C,L])
  conv_gate<<<Bm * Lm, 256, 0, stream>>>(U, w_conv, b_conv, v1, s0b);

  // 3) filter h (row 0 of implicit filter) + Toeplitz tiles
  filter_row0<<<1, 1024, 0, stream>>>(b1, w2, b2, w3, h);
  build_bt<<<33, 512, 0, stream>>>(h, bt);

  // 4) FIR via WMMA + filt_bias + s0 gate -> z bf16 [B*L, D]
  fir_gate<<<(Crows / 16) * 32, 256, 0, stream>>>(v1, s0b, bt, fb, zb);

  // 5) out = z @ w_out + b_out  (M=8192, N=1024, K=1024) -> f32 out
  dim3 g2(Mrows / BM, Dm / BN);
  gemm_bf16_tdm<float><<<g2, 256, 0, stream>>>(zb, wot, b_out, out, Mrows, Dm, Dm);
}
